// LightCurveCompressor_29798483099863
// MI455X (gfx1250) — matrix operand order repaired and checked by reading, not compiled, not run on hardware
//
#include <hip/hip_runtime.h>

// ---------------------------------------------------------------------------
// VQ-VAE vector quantizer forward (training) for MI455X / gfx1250.
// B=32768 rows, K=8192 codes, D=128 dims.
// Distance matmul + argmin fused on v_wmma_f32_16x16x32_f16 (wave32), with
// the codebook staged through LDS by the Tensor Data Mover (double-buffered).
// ---------------------------------------------------------------------------

#define VQ_B 32768
#define VQ_K 8192
#define VQ_D 128
#define VQ_DECAY 0.99f
#define VQ_COMMIT 0.25f
#define VQ_EPS 1e-5f

// LDS staging for the WMMA kernel
#define STAGE_CODES 256                      // codes per TDM stage
#define STAGES (VQ_K / STAGE_CODES)          // 32
#define TILES_PER_STAGE (STAGE_CODES / 16)   // 16 k-tiles per stage
#define STAGE_HALFS (STAGE_CODES * VQ_D)     // 32768 halfs = 64 KB
#define STAGE_BYTES (STAGE_HALFS * 2)

typedef __attribute__((ext_vector_type(16))) _Float16 v16h;
typedef __attribute__((ext_vector_type(8)))  _Float16 v8h;
typedef __attribute__((ext_vector_type(8)))  float    v8f;
typedef __attribute__((ext_vector_type(4)))  unsigned u32x4;
typedef __attribute__((ext_vector_type(8)))  int      i32x8;
typedef __attribute__((ext_vector_type(4)))  int      i32x4;

// ---------------------------------------------------------------------------
// TDM: flat 1-D copy of n_halfs f16 elements from global to LDS.
// D# layout per CDNA5 ISA 8.3/8.4: group0 = {flags, lds_addr, gaddr_lo,
// gaddr_hi|type=2}; group1 = {data_size=2B, tensor_dim0, tile_dim0, strides}.
// This toolchain's builtin takes 6 args (extra i32x8 group before cpol).
// ---------------------------------------------------------------------------
__device__ __forceinline__ void tdm_copy_to_lds(const _Float16* gsrc,
                                                unsigned lds_byte_off,
                                                unsigned n_halfs) {
  const unsigned long long ga = (unsigned long long)gsrc;
  u32x4 g0;
  g0[0] = 1u;                                          // count=1, user D#
  g0[1] = lds_byte_off;                                // lds_addr (bytes)
  g0[2] = (unsigned)(ga & 0xFFFFFFFFu);                // global_addr[31:0]
  g0[3] = (unsigned)((ga >> 32) & 0x01FFFFFFu)         // global_addr[56:32]
          | (2u << 30);                                // type = 2 ("image")
  i32x8 g1;
  g1[0] = (1 << 16);                                   // data_size=1 (2 bytes)
  g1[1] = (int)((n_halfs & 0xFFFFu) << 16);            // tensor_dim0[15:0]
  g1[2] = (int)((n_halfs >> 16) & 0xFFFFu);            // tensor_dim0[31:16]
  g1[3] = (int)((n_halfs & 0xFFFFu) << 16);            // tile_dim0 (1-D tile)
  g1[4] = 0;                                           // tile_dim1/2 unused
  g1[5] = (int)n_halfs;                                // tensor_dim0_stride lo
  g1[6] = 0;
  g1[7] = 0;
  const i32x4 z4 = {0, 0, 0, 0};                       // groups 2/3 NULL (2-D)
  const i32x8 z8 = {0, 0, 0, 0, 0, 0, 0, 0};
  __builtin_amdgcn_tensor_load_to_lds(g0, g1, z4, z4, z8, 0);
}

// ---------------------------------------------------------------------------
// K0: convert z_e and codebook to f16, zero the accumulation workspace.
// ---------------------------------------------------------------------------
__global__ void vq_prep_kernel(const float* __restrict__ ze,
                               const float* __restrict__ cb,
                               _Float16* __restrict__ zeh,
                               _Float16* __restrict__ cbh,
                               float* __restrict__ embed_sum,
                               float* __restrict__ counts,
                               float* __restrict__ scalars) {
  const int i = blockIdx.x * blockDim.x + threadIdx.x;
  if (i < VQ_B * VQ_D) zeh[i] = (_Float16)ze[i];
  if (i < VQ_K * VQ_D) { cbh[i] = (_Float16)cb[i]; embed_sum[i] = 0.0f; }
  if (i < VQ_K)        counts[i] = 0.0f;
  if (i < 2)           scalars[i] = 0.0f;   // [0]=loss_sum, [1]=n
}

// ---------------------------------------------------------------------------
// K0b: neg_half_cnorm[k] = -0.5 * |codebook[k]|^2   (one wave per code)
// ---------------------------------------------------------------------------
__global__ void vq_cnorm_kernel(const float* __restrict__ cb,
                                float* __restrict__ nhcn) {
  const int gwave = (blockIdx.x * blockDim.x + threadIdx.x) >> 5;
  const int lane  = threadIdx.x & 31;
  const float4 v = ((const float4*)(cb + (size_t)gwave * VQ_D))[lane];
  float s = v.x * v.x + v.y * v.y + v.z * v.z + v.w * v.w;
#pragma unroll
  for (int off = 16; off >= 1; off >>= 1) s += __shfl_xor(s, off, 32);
  if (lane == 0) nhcn[gwave] = -0.5f * s;
}

// ---------------------------------------------------------------------------
// K1: fused distance-GEMM + argmin.
//   dist[b,k] = |z|^2 - 2 z.w + |w|^2 ; argmin_k <=> argmax_k (z.w - 0.5|w|^2)
//   The -0.5|w|^2 term is folded into the WMMA C-matrix initializer.
// 128 threads = 4 waves; each wave owns 2 row-tiles (32 rows) of z_e kept in
// VGPRs. The f16 codebook is streamed through a 2x64KB LDS double buffer by
// the Tensor Data Mover (issued by wave 0, TENSORcnt-synced), and all 4 waves
// feed their WMMAs with ds_load_b128 reads of the shared stage.
// ---------------------------------------------------------------------------
__global__ __launch_bounds__(128) void vq_argmin_kernel(
    const _Float16* __restrict__ zeh,   // B x D (f16)
    const _Float16* __restrict__ cbh,   // K x D (f16)
    const float*    __restrict__ nhcn,  // K     (-0.5*|w|^2)
    int*            __restrict__ idx_out) {
  __shared__ _Float16 lds_cb[2 * STAGE_HALFS];   // 128 KB double buffer

  const int lane    = threadIdx.x & 31;
  const int wave    = threadIdx.x >> 5;
  const int rowbase = blockIdx.x * 128 + wave * 32;
  const int m       = lane & 15;          // A row / B,C column within tile
  const int hi      = lane >> 4;          // 0: lanes 0-15, 1: lanes 16-31
  const int klo_a   = hi * 8;             // A-matrix K sub-offset
  const int klo_b   = hi * 8;             // B-matrix K sub-offset

  // --- load A: 2 row-tiles x 4 K-chunks, resident for the whole k loop ---
  v16h a[2][4];
#pragma unroll
  for (int rt = 0; rt < 2; ++rt) {
#pragma unroll
    for (int c = 0; c < 4; ++c) {
      const _Float16* p =
          zeh + (size_t)(rowbase + rt * 16 + m) * VQ_D + c * 32 + klo_a;
      union { v16h v; v8h h[2]; } u;
      u.h[0] = *(const v8h*)(p);        // K = klo_a .. klo_a+7
      u.h[1] = *(const v8h*)(p + 16);   // K = klo_a+16 .. klo_a+23
      a[rt][c] = u.v;
    }
  }

  float mx[2][8];
  int   id[2][8];
#pragma unroll
  for (int rt = 0; rt < 2; ++rt)
#pragma unroll
    for (int r = 0; r < 8; ++r) { mx[rt][r] = -__builtin_inff(); id[rt][r] = 0; }

  // LDS aperture keeps the byte offset in addr[31:0] of the generic pointer.
  const unsigned lds_base = (unsigned)(unsigned long long)(const void*)&lds_cb[0];

  // prologue: stage 0 into buffer 0
  if (wave == 0) {
    tdm_copy_to_lds(cbh, lds_base, STAGE_HALFS);
    __builtin_amdgcn_s_wait_tensorcnt(0);
  }
  __syncthreads();

  for (int s = 0; s < STAGES; ++s) {
    const _Float16* lbuf = &lds_cb[(s & 1) * STAGE_HALFS];
    if (wave == 0) {
      if (s + 1 < STAGES) {
        tdm_copy_to_lds(cbh + (size_t)(s + 1) * STAGE_HALFS,
                        lds_base + (unsigned)(((s + 1) & 1) * STAGE_BYTES),
                        STAGE_HALFS);
        __builtin_prefetch(nhcn + (s + 1) * STAGE_CODES, 0, 1);
      }
    }

    for (int t = 0; t < TILES_PER_STAGE; ++t) {
      const int kt  = s * TILES_PER_STAGE + t;
      const int col = kt * 16 + m;          // code index this lane scores
      const float nh = nhcn[col];
      v8f acc0, acc1;
#pragma unroll
      for (int r = 0; r < 8; ++r) { acc0[r] = nh; acc1[r] = nh; }

      const _Float16* bp = lbuf + (t * 16 + m) * VQ_D + klo_b;
#pragma unroll
      for (int c = 0; c < 4; ++c) {
        union { v16h v; v8h h[2]; } ub;
        ub.h[0] = *(const v8h*)(bp + c * 32);       // K = klo_b .. klo_b+7
        ub.h[1] = *(const v8h*)(bp + c * 32 + 16);  // K = klo_b+16 .. klo_b+23
        const v16h b = ub.v;
        acc0 = __builtin_amdgcn_wmma_f32_16x16x32_f16(
            false, a[0][c], false, b, (short)0, acc0, false, false);
        acc1 = __builtin_amdgcn_wmma_f32_16x16x32_f16(
            false, a[1][c], false, b, (short)0, acc1, false, false);
      }
#pragma unroll
      for (int r = 0; r < 8; ++r) {
        if (acc0[r] > mx[0][r]) { mx[0][r] = acc0[r]; id[0][r] = col; }
        if (acc1[r] > mx[1][r]) { mx[1][r] = acc1[r]; id[1][r] = col; }
      }
    }

    // next stage's DMA must be complete before any wave touches that buffer
    if (wave == 0) __builtin_amdgcn_s_wait_tensorcnt(0);
    __syncthreads();
  }

  // --- cross-lane argmax within each 16-lane half (columns of one row) ---
#pragma unroll
  for (int rt = 0; rt < 2; ++rt) {
#pragma unroll
    for (int r = 0; r < 8; ++r) {
      float v = mx[rt][r];
      int   i = id[rt][r];
#pragma unroll
      for (int off = 8; off >= 1; off >>= 1) {
        const float ov = __shfl_xor(v, off, 32);
        const int   oi = __shfl_xor(i, off, 32);
        if (ov > v || (ov == v && oi < i)) { v = ov; i = oi; }
      }
      // row M = r (lanes 0-15) or r+8 (lanes 16-31)
      if (m == 0) idx_out[rowbase + rt * 16 + hi * 8 + r] = i;
    }
  }
}

// ---------------------------------------------------------------------------
// K2: per-row gather + scatter. One 128-thread block per row b.
//   z_q_st = codebook[idx[b]] ; loss += |z_q - z_e|^2 ;
//   embed_sum[idx[b]] += z_e[b] ; counts[idx[b]] += 1 ; indices -> float out
// ---------------------------------------------------------------------------
__global__ __launch_bounds__(128) void vq_scatter_kernel(
    const float* __restrict__ ze, const float* __restrict__ cb,
    const int* __restrict__ idx, float* __restrict__ out_zq,
    float* __restrict__ out_idx_f, float* __restrict__ embed_sum,
    float* __restrict__ counts, float* __restrict__ loss_sum) {
  const int b = blockIdx.x;
  const int d = threadIdx.x;
  const int k = idx[b];
  const float zq = cb[(size_t)k * VQ_D + d];
  const float zv = ze[(size_t)b * VQ_D + d];
  out_zq[(size_t)b * VQ_D + d] = zq;           // straight-through value == z_q
  atomicAdd(&embed_sum[(size_t)k * VQ_D + d], zv);
  const float diff = zq - zv;
  float s = diff * diff;
#pragma unroll
  for (int off = 16; off >= 1; off >>= 1) s += __shfl_xor(s, off, 32);
  __shared__ float red[4];
  if ((threadIdx.x & 31) == 0) red[threadIdx.x >> 5] = s;
  __syncthreads();
  if (threadIdx.x == 0) {
    atomicAdd(loss_sum, red[0] + red[1] + red[2] + red[3]);
    atomicAdd(&counts[k], 1.0f);
    out_idx_f[b] = (float)k;
  }
}

// ---------------------------------------------------------------------------
// K3a: new_cluster_size = decay*ema + (1-decay)*counts ; n = sum(new_cs)
// ---------------------------------------------------------------------------
__global__ __launch_bounds__(256) void vq_ema_cs_kernel(
    const float* __restrict__ ema_cs, const float* __restrict__ counts,
    float* __restrict__ out_ncs, float* __restrict__ n_acc) {
  const int k = blockIdx.x * 256 + threadIdx.x;
  const float ncs = VQ_DECAY * ema_cs[k] + (1.0f - VQ_DECAY) * counts[k];
  out_ncs[k] = ncs;
  float s = ncs;
#pragma unroll
  for (int off = 16; off >= 1; off >>= 1) s += __shfl_xor(s, off, 32);
  __shared__ float red[8];
  if ((threadIdx.x & 31) == 0) red[threadIdx.x >> 5] = s;
  __syncthreads();
  if (threadIdx.x == 0) {
    float t = 0.0f;
#pragma unroll
    for (int w = 0; w < 8; ++w) t += red[w];
    atomicAdd(n_acc, t);
  }
}

// ---------------------------------------------------------------------------
// K3b: new_embed_sum, new_codebook, vq_loss scalar.
// ---------------------------------------------------------------------------
__global__ __launch_bounds__(256) void vq_finalize_kernel(
    const float* __restrict__ ema_es, const float* __restrict__ embed_sum,
    const float* __restrict__ ncs, const float* __restrict__ scalars,
    float* __restrict__ out_nes, float* __restrict__ out_ncb,
    float* __restrict__ out_loss) {
  const int i = blockIdx.x * 256 + threadIdx.x;   // < K*D
  const int k = i >> 7;
  const float n = scalars[1];
  const float cs = (ncs[k] + VQ_EPS) / (n + (float)VQ_K * VQ_EPS) * n;
  const float nes = VQ_DECAY * ema_es[i] + (1.0f - VQ_DECAY) * embed_sum[i];
  out_nes[i] = nes;
  out_ncb[i] = nes / cs;
  if (i == 0)
    *out_loss = VQ_COMMIT * scalars[0] / (float)((size_t)VQ_B * VQ_D);
}

// ---------------------------------------------------------------------------
// Launch
// ---------------------------------------------------------------------------
extern "C" void kernel_launch(void* const* d_in, const int* in_sizes, int n_in,
                              void* d_out, int out_size, void* d_ws,
                              size_t ws_size, hipStream_t stream) {
  const float* z_e      = (const float*)d_in[0];   // B*D
  const float* codebook = (const float*)d_in[1];   // K*D
  const float* ema_cs   = (const float*)d_in[2];   // K
  const float* ema_es   = (const float*)d_in[3];   // K*D
  float* out = (float*)d_out;

  // output layout (floats), concatenated in reference return order
  float* o_zq   = out;                                      // B*D
  float* o_loss = out + (size_t)VQ_B * VQ_D;                // 1
  float* o_idx  = o_loss + 1;                               // B
  float* o_ncb  = o_idx + VQ_B;                             // K*D
  float* o_ncs  = o_ncb + (size_t)VQ_K * VQ_D;              // K
  float* o_nes  = o_ncs + VQ_K;                             // K*D

  // workspace layout
  char* ws = (char*)d_ws;
  _Float16* zeh    = (_Float16*)(ws);                                  // 8 MB
  _Float16* cbh    = (_Float16*)(ws + (size_t)VQ_B * VQ_D * 2);        // 2 MB
  float* nhcn      = (float*)(ws + (size_t)(VQ_B + VQ_K) * VQ_D * 2);  // 32 KB
  float* idxws     = nhcn + VQ_K;                                      // 128 KB
  int*   idx_i     = (int*)idxws;
  float* counts    = idxws + VQ_B;                                     // 32 KB
  float* embed_sum = counts + VQ_K;                                    // 4 MB
  float* scalars   = embed_sum + (size_t)VQ_K * VQ_D;                  // 8 B

  // K0: convert + zero accumulators
  vq_prep_kernel<<<(VQ_B * VQ_D + 255) / 256, 256, 0, stream>>>(
      z_e, codebook, zeh, cbh, embed_sum, counts, scalars);

  // K0b: code norms (8 codes per 256-thread block)
  vq_cnorm_kernel<<<VQ_K / 8, 256, 0, stream>>>(codebook, nhcn);

  // K1: WMMA distance + argmin (128 rows per block, TDM-staged codebook)
  vq_argmin_kernel<<<VQ_B / 128, 128, 0, stream>>>(zeh, cbh, nhcn, idx_i);

  // K2: gather/scatter per row
  vq_scatter_kernel<<<VQ_B, 128, 0, stream>>>(
      z_e, codebook, idx_i, o_zq, o_idx, embed_sum, counts, scalars);

  // K3a: EMA cluster size + n
  vq_ema_cs_kernel<<<VQ_K / 256, 256, 0, stream>>>(ema_cs, counts, o_ncs,
                                                   scalars + 1);

  // K3b: EMA embed sum, normalized codebook, loss scalar
  vq_finalize_kernel<<<VQ_K * VQ_D / 256, 256, 0, stream>>>(
      ema_es, embed_sum, o_ncs, scalars, o_nes, o_ncb, o_loss);
}
